// SpatialModulatedConv2d_76948634075555
// MI455X (gfx1250) — matrix operand order repaired and checked by reading, not compiled, not run on hardware
//
#include <hip/hip_runtime.h>

// ---------------------------------------------------------------------------
// Types / helpers for CDNA5 WMMA (wave32, 16x16x32 bf16 -> f32)
// ---------------------------------------------------------------------------
typedef __attribute__((ext_vector_type(16))) __bf16 bf16x16;
typedef __attribute__((ext_vector_type(8)))  float  floatx8;

union Frag {
  bf16x16 bf;
  unsigned short us[16];
  uint4 q[2];
};

__device__ __forceinline__ unsigned short f2bf(float f) {
  union { float f; unsigned int u; } c; c.f = f;
  unsigned int u = c.u;
  return (unsigned short)((u + 0x7FFFu + ((u >> 16) & 1u)) >> 16);
}

__device__ __forceinline__ floatx8 wmma_bf16(const Frag& a, const Frag& b, floatx8 c) {
  // 8 args: (neg_a, A, neg_b, B, c_mod, C, reuse_a, reuse_b)
  return __builtin_amdgcn_wmma_f32_16x16x32_bf16(false, a.bf, false, b.bf,
                                                 (short)0, c, false, false);
}

__device__ __forceinline__ void loadB(Frag& B, const unsigned short* p) {
  B.q[0] = *(const uint4*)p;
  B.q[1] = *(const uint4*)(p + 8);
}

__device__ __forceinline__ floatx8 fzero8() {
  floatx8 z = {0.f,0.f,0.f,0.f,0.f,0.f,0.f,0.f};
  return z;
}

#define SCALE_MAIN 0.029462782549439483f   // 1/sqrt(128*9)
#define LSCALE     0.044194173824159216f   // 1/sqrt(512)
#define SQRT2      1.41421356237309515f

// ---------------------------------------------------------------------------
// Weff[ph(4)][tap(9)][cout(128)][cin(128)] bf16
// Fused (convT stride2, 3x3, flipped) o (4x4 separable blur) == per-phase 3x3
// conv over the ORIGINAL 128x128 grid. dy = 2t+1-py-ky selects blur row tap.
// ---------------------------------------------------------------------------
__global__ __launch_bounds__(256) void prep_weff(const float* __restrict__ w,
                                                 unsigned short* __restrict__ weff) {
  int idx = blockIdx.x * 256 + threadIdx.x;
  if (idx >= 4 * 9 * 128 * 128) return;
  int cin  = idx & 127;
  int cout = (idx >> 7) & 127;
  int pt   = idx >> 14;          // ph*9 + tap
  int tap  = pt % 9;
  int ph   = pt / 9;
  int t = tap / 3, s = tap % 3;
  int py = ph >> 1, px = ph & 1;
  const float kk[4] = {1.f, 3.f, 3.f, 1.f};
  float acc = 0.f;
  #pragma unroll
  for (int ky = 0; ky < 3; ++ky) {
    int dy = 2 * t + 1 - py - ky;
    if (dy < 0 || dy > 3) continue;
    float ay = kk[3 - dy] * 0.25f;
    #pragma unroll
    for (int kx = 0; kx < 3; ++kx) {
      int dx = 2 * s + 1 - px - kx;
      if (dx < 0 || dx > 3) continue;
      float ax = kk[3 - dx] * 0.25f;
      // wf[ky][kx] = scale * w[2-ky][2-kx]
      acc += ay * ax * w[((cout * 128 + cin) * 3 + (2 - ky)) * 3 + (2 - kx)];
    }
  }
  weff[idx] = f2bf(acc * SCALE_MAIN);
}

// W2[conv(2)][tap(9)][cout(128)][cin(128)] bf16 (plain correlation taps)
__global__ __launch_bounds__(256) void prep_w2(const float* __restrict__ convw_w,
                                               const float* __restrict__ convb_w,
                                               unsigned short* __restrict__ w2) {
  int idx = blockIdx.x * 256 + threadIdx.x;
  if (idx >= 2 * 9 * 128 * 128) return;
  int cin  = idx & 127;
  int cout = (idx >> 7) & 127;
  int ct   = idx >> 14;          // conv*9 + tap
  int tap  = ct % 9;
  int conv = ct / 9;
  const float* src = conv ? convb_w : convw_w;
  w2[idx] = f2bf(src[(cout * 128 + cin) * 9 + tap] * SCALE_MAIN);
}

__global__ void zero_stats(float* s) {
  int i = blockIdx.x * 256 + threadIdx.x;
  if (i < 8 * 256) s[i] = 0.f;
}

// ---------------------------------------------------------------------------
// Main fused kernel: convT(stride2)+blur as 4-phase 3x3 implicit GEMM.
// Block = (b, ys, 32-col x-tile); 8 waves: wave -> (phase = w&3, mtile = w>>2).
// Each wave accumulates 8 cout-tiles (16x16) over K = 9 taps * 128 cin.
// B-fragment stream is software-pipelined (Bcur/Bnext) across the whole
// 9*4*8 iteration space so every WMMA overlaps the next load's latency.
// ---------------------------------------------------------------------------
#define XT_STRIDE 136   // 128 cin + 8 pad -> conflict-free ds_read_b128
__global__ __launch_bounds__(256)
void fused_convT_blur(const unsigned short* __restrict__ weff,
                      const float* __restrict__ x,
                      float* __restrict__ blur,
                      float* __restrict__ stats) {
  __shared__ unsigned short xtile[3 * 34 * XT_STRIDE];  // 27744 B, bf16 [t][col][cin]
  __shared__ float statsLDS[256];

  int bid = blockIdx.x;
  int xt  = bid & 3;
  int ys  = (bid >> 2) & 127;
  int b   = bid >> 9;
  int xs0 = xt * 32;
  int tid = threadIdx.x;

  statsLDS[tid] = 0.f;

  // Stage x neighborhood: rows ys-1..ys+1, cols xs0-1..xs0+32, all 128 cin, bf16.
  for (int i = tid; i < 3 * 34 * 128; i += 256) {
    int col = i % 34;
    int cin = (i / 34) & 127;
    int t   = i / (34 * 128);
    int r = ys + t - 1;
    int c = xs0 + col - 1;
    float v = (r >= 0 && r < 128 && c >= 0 && c < 128)
                ? x[(((size_t)b * 128 + cin) * 128 + r) * 128 + c] : 0.f;
    xtile[(t * 34 + col) * XT_STRIDE + cin] = f2bf(v);
  }
  __syncthreads();

  int wave = tid >> 5, lane = tid & 31;
  int ph = wave & 3, mt = wave >> 2;
  int py = ph >> 1, px = ph & 1;
  int ln = lane & 15, lh = lane >> 4;

  floatx8 acc[8];
  #pragma unroll
  for (int ct = 0; ct < 8; ++ct) acc[ct] = fzero8();

  const unsigned short* wph = weff + (size_t)ph * 9 * 16384;
  const int boff = ln * 128 + 16 * lh;          // + ct*2048 + kc*32

  Frag Bc;
  loadB(Bc, wph + boff);                        // prime pipeline: (tap0, kc0, ct0)

  for (int tap = 0; tap < 9; ++tap) {
    int t = tap / 3, s = tap % 3;
    const unsigned short* wtap = wph + tap * 16384;
    const unsigned short* wnxt = wph + (tap < 8 ? tap + 1 : tap) * 16384;
    // Warm next tap's weight slab in L0 (global_prefetch_b8).
    __builtin_prefetch((const void*)(wnxt + ((mt * 32 + lane) * 64)), 0, 1);

    // Hoist all 4 A-fragments for this tap (8x ds_read_b128).
    const unsigned short* ap = &xtile[(t * 34 + mt * 16 + ln + s) * XT_STRIDE];
    Frag A4[4];
    #pragma unroll
    for (int kc = 0; kc < 4; ++kc) {
      A4[kc].q[0] = *(const uint4*)(ap + kc * 32 + 16 * lh);
      A4[kc].q[1] = *(const uint4*)(ap + kc * 32 + 16 * lh + 8);
    }

    #pragma unroll
    for (int kc = 0; kc < 4; ++kc) {
      #pragma unroll
      for (int ct = 0; ct < 8; ++ct) {
        const unsigned short* np;
        if (ct < 7)      np = wtap + boff + kc * 32 + (ct + 1) * 2048;
        else if (kc < 3) np = wtap + boff + (kc + 1) * 32;
        else             np = wnxt + boff;       // first B of next tap
        Frag Bn;
        loadB(Bn, np);
        acc[ct] = wmma_bf16(A4[kc], Bc, acc[ct]);
        Bc = Bn;                                 // SSA-renamed away when unrolled
      }
    }
  }

  // Epilogue: store un-normalized result + accumulate per-channel stats.
  int y = 2 * ys + py;
  #pragma unroll
  for (int ct = 0; ct < 8; ++ct) {
    int cout = ct * 16 + ln;
    float s1 = 0.f, s2 = 0.f;
    #pragma unroll
    for (int r = 0; r < 8; ++r) {
      int m = r + 8 * lh;                        // C/D: VGPR r -> M = r + 8*half
      int xc = 2 * (xs0 + mt * 16 + m) + px;
      float v = acc[ct][r];
      blur[(((size_t)b * 128 + cout) * 256 + y) * 256 + xc] = v;
      s1 += v;
      s2 += v * v;
    }
    atomicAdd(&statsLDS[cout], s1);
    atomicAdd(&statsLDS[128 + cout], s2);
  }
  __syncthreads();
  atomicAdd(&stats[b * 256 + tid], statsLDS[tid]);
}

// ---------------------------------------------------------------------------
// Style EqualLinear: s = leaky(style @ (lin_w*ls)^T + b, 0.2)*sqrt(2)
// GEMM M=8(pad16), N=32768, K=512 -> smap bf16 laid out [b][iy][ix][cin].
// ---------------------------------------------------------------------------
__global__ __launch_bounds__(256)
void style_linear(const float* __restrict__ style,
                  const float* __restrict__ lin_w,
                  const float* __restrict__ lin_b,
                  unsigned short* __restrict__ smap) {
  __shared__ unsigned short sty[16 * 512];
  int tid = threadIdx.x;
  for (int i = tid; i < 16 * 512; i += 256) {
    int m = i >> 9, k = i & 511;
    sty[i] = f2bf(m < 8 ? style[m * 512 + k] : 0.f);
  }
  __syncthreads();

  int wave = tid >> 5, lane = tid & 31;
  int ln = lane & 15, lh = lane >> 4;
  int n0 = (blockIdx.x * 8 + wave) * 16;

  floatx8 acc = fzero8();
  const unsigned short* ap = &sty[ln * 512];
  for (int kc = 0; kc < 16; ++kc) {
    int c0 = kc * 32;
    Frag A;
    A.q[0] = *(const uint4*)(ap + c0 + 16 * lh);
    A.q[1] = *(const uint4*)(ap + c0 + 16 * lh + 8);
    const float* bp = lin_w + (size_t)(n0 + ln) * 512 + c0 + 16 * lh;
    Frag B;
    #pragma unroll
    for (int e = 0; e < 16; ++e) B.us[e] = f2bf(bp[e] * LSCALE);
    acc = wmma_bf16(A, B, acc);
  }

  int nout = n0 + ln;
  float bias = lin_b[nout];
  int cin = nout >> 8, iy = (nout >> 4) & 15, ix = nout & 15;
  #pragma unroll
  for (int r = 0; r < 8; ++r) {
    int m = r + 8 * lh;                            // m = batch
    if (m < 8) {
      float v = acc[r] + bias;
      v = (v > 0.f ? v : 0.2f * v) * SQRT2;
      smap[(((size_t)m * 16 + iy) * 16 + ix) * 128 + cin] = f2bf(v);
    }
  }
}

// ---------------------------------------------------------------------------
// Two EqualConv2d(128,128,3,pad=1) on the x2-nearest-upsampled 32x32 style map,
// upsample folded into A addressing. Output [conv][b][32][32][cout] fp32.
// Block = (b, y); 8 waves -> (conv, xtile, cout-half). B stream pipelined.
// ---------------------------------------------------------------------------
__global__ __launch_bounds__(256)
void mod_convs(const unsigned short* __restrict__ smap,
               const unsigned short* __restrict__ w2,
               const float* __restrict__ convw_b,
               const float* __restrict__ convb_b,
               float* __restrict__ swsb) {
  int bid = blockIdx.x;
  int y = bid & 31;
  int b = bid >> 5;
  int tid = threadIdx.x, wave = tid >> 5, lane = tid & 31;
  int conv = wave & 1, xt = (wave >> 1) & 1, chalf = wave >> 2;
  int ln = lane & 15, lh = lane >> 4;

  floatx8 acc[4];
  #pragma unroll
  for (int i = 0; i < 4; ++i) acc[i] = fzero8();

  const unsigned short* wcv = w2 + (size_t)conv * 9 * 16384;
  const int boff = ln * 128 + 16 * lh + chalf * 4 * 2048;

  Frag Bc;
  loadB(Bc, wcv + boff);                           // (tap0, kc0, i0)

  for (int tap = 0; tap < 9; ++tap) {
    int ky = tap / 3, kx = tap % 3;
    const unsigned short* wtap = wcv + tap * 16384;
    const unsigned short* wnxt = wcv + (tap < 8 ? tap + 1 : tap) * 16384;

    int iy_up = y + ky - 1;
    int xpos = xt * 16 + ln;
    int ix_up = xpos + kx - 1;
    bool ok = (iy_up >= 0 && iy_up < 32 && ix_up >= 0 && ix_up < 32);
    int iy = ok ? (iy_up >> 1) : 0;
    int ix = ok ? (ix_up >> 1) : 0;
    const unsigned short* ap = smap + (((size_t)b * 16 + iy) * 16 + ix) * 128;

    Frag A4[4];
    #pragma unroll
    for (int kc = 0; kc < 4; ++kc) {
      if (ok) {
        A4[kc].q[0] = *(const uint4*)(ap + kc * 32 + 16 * lh);
        A4[kc].q[1] = *(const uint4*)(ap + kc * 32 + 16 * lh + 8);
      } else {
        A4[kc].q[0] = make_uint4(0u, 0u, 0u, 0u);
        A4[kc].q[1] = make_uint4(0u, 0u, 0u, 0u);
      }
    }

    #pragma unroll
    for (int kc = 0; kc < 4; ++kc) {
      #pragma unroll
      for (int i = 0; i < 4; ++i) {
        const unsigned short* np;
        if (i < 3)       np = wtap + boff + kc * 32 + (i + 1) * 2048;
        else if (kc < 3) np = wtap + boff + (kc + 1) * 32;
        else             np = wnxt + boff;
        Frag Bn;
        loadB(Bn, np);
        acc[i] = wmma_bf16(A4[kc], Bc, acc[i]);
        Bc = Bn;
      }
    }
  }

  const float* biasar = conv ? convb_b : convw_b;
  #pragma unroll
  for (int i = 0; i < 4; ++i) {
    int cout = (chalf * 4 + i) * 16 + ln;
    float bias = biasar[cout];
    #pragma unroll
    for (int r = 0; r < 8; ++r) {
      int m = r + 8 * lh;
      int xpos = xt * 16 + m;
      swsb[((((size_t)conv * 8 + b) * 32 + y) * 32 + xpos) * 128 + cout] = acc[i][r] + bias;
    }
  }
}

// ---------------------------------------------------------------------------
// Finalize: InstanceNorm (from accumulated stats) + spatial modulation.
// ---------------------------------------------------------------------------
__global__ __launch_bounds__(256)
void finalize(const float* __restrict__ blur,
              const float* __restrict__ stats,
              const float* __restrict__ swsb,
              float* __restrict__ out) {
  int plane = blockIdx.x >> 6;                     // b*128 + c
  int b = plane >> 7, c = plane & 127;
  float sum = stats[b * 256 + c];
  float sq  = stats[b * 256 + 128 + c];
  float mean = sum * (1.f / 65536.f);
  float var  = sq * (1.f / 65536.f) - mean * mean;
  float inv  = rsqrtf(var + 1e-5f);

  int off = ((blockIdx.x & 63) << 10) + (threadIdx.x << 2);  // within 256x256 plane
  int y = off >> 8, xc = off & 255;
  int sy = y >> 3, sx = xc >> 3;                   // nearest down-map 256 -> 32
  size_t sidx = (((size_t)b * 32 + sy) * 32 + sx) * 128 + c;
  float sw = swsb[sidx];
  float sb = swsb[(size_t)8 * 32 * 32 * 128 + sidx];
  float ga = 1.f + 0.3f * sw;
  float gb = 0.3f * sb;

  size_t base = (size_t)plane * 65536 + off;
  float4 v = *(const float4*)(blur + base);
  float4 o;
  o.x = (v.x - mean) * inv * ga + gb;
  o.y = (v.y - mean) * inv * ga + gb;
  o.z = (v.z - mean) * inv * ga + gb;
  o.w = (v.w - mean) * inv * ga + gb;
  *(float4*)(out + base) = o;
}

// ---------------------------------------------------------------------------
extern "C" void kernel_launch(void* const* d_in, const int* in_sizes, int n_in,
                              void* d_out, int out_size, void* d_ws, size_t ws_size,
                              hipStream_t stream) {
  const float* x       = (const float*)d_in[0];
  const float* style   = (const float*)d_in[1];
  const float* weight  = (const float*)d_in[2];
  const float* lin_w   = (const float*)d_in[3];
  const float* lin_b   = (const float*)d_in[4];
  const float* convw_w = (const float*)d_in[5];
  const float* convw_b = (const float*)d_in[6];
  const float* convb_w = (const float*)d_in[7];
  const float* convb_b = (const float*)d_in[8];
  float* out = (float*)d_out;

  char* ws = (char*)d_ws;
  float* blurbuf        = (float*)(ws);                          // 268,435,456 B
  unsigned short* weff  = (unsigned short*)(ws + 268435456);     //   1,179,648 B
  unsigned short* w2    = (unsigned short*)(ws + 269615104);     //     589,824 B
  float* stats          = (float*)(ws + 270204928);              //       8,192 B
  unsigned short* smap  = (unsigned short*)(ws + 270213120);     //     524,288 B
  float* swsb           = (float*)(ws + 270737408);              //   8,388,608 B

  prep_weff<<<(4 * 9 * 128 * 128 + 255) / 256, 256, 0, stream>>>(weight, weff);
  prep_w2<<<(2 * 9 * 128 * 128 + 255) / 256, 256, 0, stream>>>(convw_w, convb_w, w2);
  zero_stats<<<8, 256, 0, stream>>>(stats);

  fused_convT_blur<<<8 * 128 * 4, 256, 0, stream>>>(weff, x, blurbuf, stats);

  style_linear<<<256, 256, 0, stream>>>(style, lin_w, lin_b, smap);
  mod_convs<<<8 * 32, 256, 0, stream>>>(smap, w2, convw_b, convb_b, swsb);

  finalize<<<8 * 128 * 64, 256, 0, stream>>>(blurbuf, stats, swsb, out);
}
